// Model_41575283425320
// MI455X (gfx1250) — matrix operand order repaired and checked by reading, not compiled, not run on hardware
//
#include <hip/hip_runtime.h>

// ---------------------------------------------------------------------------
// Model constants (match the JAX reference)
// ---------------------------------------------------------------------------
#define BATCH 32
#define SEQL  512      // sequence length inside mambas ( == C == NEMB )
#define CH    512      // channels C
#define NEMBD 512      // NEMB
#define PRED  96
#define DIN   1024     // d_inner = EXPAND * d_model
#define DTR   32       // dt_rank = ceil(512/16)
#define NST   16       // DSTATE

typedef unsigned short ushort_t;
typedef __attribute__((ext_vector_type(16))) __bf16 v16bf;
typedef __attribute__((ext_vector_type(8)))  float  v8f;

struct __attribute__((aligned(16))) U4 { unsigned int x, y, z, w; };
union Frag { U4 u[2]; v16bf v; };

__device__ __forceinline__ ushort_t f2bf(float f) {
  union { float f; unsigned int u; } c; c.f = f;
  unsigned int r = c.u + 0x7FFFu + ((c.u >> 16) & 1u);   // round-to-nearest-even
  return (ushort_t)(r >> 16);
}
__device__ __forceinline__ float bf2f(ushort_t h) {
  union { unsigned int u; float f; } c; c.u = ((unsigned int)h) << 16;
  return c.f;
}
__device__ __forceinline__ float silu_f(float x) { return x / (1.f + __expf(-x)); }

// ---------------------------------------------------------------------------
// Generic batched bf16 GEMM with WMMA:  Out[b] = A[b] (MxK) * W^T (NxK) + bias
// A, W row-major bf16.  Out fp32 or bf16.  postop 1 = softplus.
// Block: 256 threads (8 waves), tile M=128, N=64, K-step 32.
// Staging: GLOBAL_LOAD_ASYNC_TO_LDS_B128 (ASYNCcnt), double-buffered LDS so
// next tile's global->LDS DMA runs under the current tile's WMMAs.
// ---------------------------------------------------------------------------
#define TM 128
#define TN 64
#define TK 32
#define LROW 48   // halfs per LDS row (96B: keeps 16B alignment, breaks pow2 conflicts)

#define ASYNC_B128(ldsoff, gaddr) \
  asm volatile("global_load_async_to_lds_b128 %0, %1, off" :: "v"(ldsoff), "v"(gaddr) : "memory")
#define WAIT_ASYNC0() asm volatile("s_wait_asynccnt 0x0" ::: "memory")

__global__ void __launch_bounds__(256) gemm_bf16_kernel(
    const ushort_t* __restrict__ A, long long strideA, int lda,
    const ushort_t* __restrict__ W,
    void* __restrict__ Out, long long strideO, int ldo,
    int N, int K,
    const float* __restrict__ bias, int postop, int out_bf16)
{
  __shared__ ushort_t As[2][TM * LROW];
  __shared__ ushort_t Ws[2][TN * LROW];

  const int tid  = threadIdx.x;
  const int lane = tid & 31;
  const int wave = tid >> 5;
  const int wm = wave >> 1;           // 0..3  (rows of 32)
  const int wn = wave & 1;            // 0..1  (cols of 32)
  const int mblock = blockIdx.y * TM; // M is always a multiple of TM in this model
  const int nblock = blockIdx.x * TN;
  const ushort_t* Ab = A + (size_t)blockIdx.z * strideA;
  const int lh  = lane >> 4;          // half-wave select
  const int l15 = lane & 15;

  // ---- per-thread staging assignment: three 16B chunks per K-step ----
  const int srow = tid >> 2;          // 0..63
  const int skc  = tid & 3;           // 0..3  (which 8-half chunk of the 32-half row)
  unsigned long long gA0 = (unsigned long long)(Ab + (size_t)(mblock + srow) * lda + skc * 8);
  unsigned long long gA1 = (unsigned long long)(Ab + (size_t)(mblock + srow + 64) * lda + skc * 8);
  int nrow = nblock + srow;
  if (nrow >= N) nrow = N - 1;        // clamp: junk cols are masked in the epilogue
  unsigned long long gW = (unsigned long long)(W + (size_t)nrow * K + skc * 8);

  // LDS byte offsets (flat shared address truncated to the 32-bit LDS offset)
  unsigned ldsA0[2], ldsA1[2], ldsW0[2];
#pragma unroll
  for (int p = 0; p < 2; ++p) {
    ldsA0[p] = (unsigned)(size_t)&As[p][srow * LROW + skc * 8];
    ldsA1[p] = (unsigned)(size_t)&As[p][(srow + 64) * LROW + skc * 8];
    ldsW0[p] = (unsigned)(size_t)&Ws[p][srow * LROW + skc * 8];
  }

  v8f vz = {0.f,0.f,0.f,0.f,0.f,0.f,0.f,0.f};
  v8f acc[2][2];
  acc[0][0] = vz; acc[0][1] = vz; acc[1][0] = vz; acc[1][1] = vz;

  // prologue: DMA tile 0 into buffer 0
  ASYNC_B128(ldsA0[0], gA0);
  ASYNC_B128(ldsA1[0], gA1);
  ASYNC_B128(ldsW0[0], gW);

  const int nk = K / TK;
  for (int i = 0; i < nk; ++i) {
    const int p = i & 1;
    WAIT_ASYNC0();        // own chunks of tile i landed in LDS
    __syncthreads();      // all waves' chunks visible; other buffer no longer read
    if (i + 1 < nk) {     // DMA tile i+1 into the other buffer, runs under WMMAs
      gA0 += 2 * TK; gA1 += 2 * TK; gW += 2 * TK;   // advance 64 bytes
      ASYNC_B128(ldsA0[p ^ 1], gA0);
      ASYNC_B128(ldsA1[p ^ 1], gA1);
      ASYNC_B128(ldsW0[p ^ 1], gW);
    }

    const ushort_t* Ap = As[p];
    const ushort_t* Wp = Ws[p];
    // A fragment: lane = M row (l15), halves hold K {0..7,16..23} / {8..15,24..31}
    Frag a[2], fb[2];
#pragma unroll
    for (int r = 0; r < 2; ++r) {
      int row = wm * 32 + r * 16 + l15;
      int kb  = lh * 8;
      a[r].u[0] = *(const U4*)&Ap[row * LROW + kb];
      a[r].u[1] = *(const U4*)&Ap[row * LROW + kb + 16];
    }
    // B fragment: lane = N col (l15), halves hold K {0..7,16..23} / {8..15,24..31}
#pragma unroll
    for (int c = 0; c < 2; ++c) {
      int nr = wn * 32 + c * 16 + l15;
      int kb = lh * 8;
      fb[c].u[0] = *(const U4*)&Wp[nr * LROW + kb];
      fb[c].u[1] = *(const U4*)&Wp[nr * LROW + kb + 16];
    }
#pragma unroll
    for (int r = 0; r < 2; ++r)
#pragma unroll
      for (int c = 0; c < 2; ++c)
        acc[r][c] = __builtin_amdgcn_wmma_f32_16x16x32_bf16(
            false, a[r].v, false, fb[c].v, (short)0, acc[r][c], false, false);
  }

  // epilogue: C/D layout -> row = i + 8*lh, col = l15 within each 16x16 tile
#pragma unroll
  for (int r = 0; r < 2; ++r)
    for (int c = 0; c < 2; ++c) {
      int colg = nblock + wn * 32 + c * 16 + l15;
      if (colg >= N) continue;
      float bv = bias ? bias[colg] : 0.f;
#pragma unroll
      for (int i = 0; i < 8; ++i) {
        int rowg = mblock + wm * 32 + r * 16 + i + lh * 8;
        float v = acc[r][c][i] + bv;
        if (postop == 1) v = (v > 20.f) ? v : __logf(1.f + __expf(v));   // softplus
        size_t oidx = (size_t)blockIdx.z * strideO + (size_t)rowg * ldo + colg;
        if (out_bf16) ((ushort_t*)Out)[oidx] = f2bf(v);
        else          ((float*)Out)[oidx]    = v;
      }
    }
}

// ---------------------------------------------------------------------------
// fp32 -> bf16 weight/activation conversion
// ---------------------------------------------------------------------------
__global__ void cvt_kernel(const float* __restrict__ in, ushort_t* __restrict__ out, int n) {
  int i = blockIdx.x * 256 + threadIdx.x;
  if (i < n) out[i] = f2bf(in[i]);
}

// ---------------------------------------------------------------------------
// RevIN statistics: mean / 1/std over L per (b,c)
// ---------------------------------------------------------------------------
__global__ void stats_kernel(const float* __restrict__ x,
                             float* __restrict__ mu, float* __restrict__ rstd) {
  int t = threadIdx.x;
  int c = ((blockIdx.x & 1) << 8) + t;
  int b = blockIdx.x >> 1;
  float s = 0.f, ss = 0.f;
  for (int l = 0; l < SEQL; ++l) {
    float v = x[(size_t)(b * SEQL + l) * CH + c];
    s += v; ss += v * v;
  }
  float m   = s * (1.f / SEQL);
  float var = ss * (1.f / SEQL) - m * m;
  mu[b * CH + c]   = m;
  rstd[b * CH + c] = rsqrtf(var + 1e-5f);
}

// ---------------------------------------------------------------------------
// Normalize + 25-tap edge-replicated moving average; write seas|trend
// transposed into xcatT[b][c][0..1023] (bf16) so lin2 is a plain GEMM.
// ---------------------------------------------------------------------------
__global__ void decomp_kernel(const float* __restrict__ x, const float* __restrict__ mu,
                              const float* __restrict__ rstd, const float* __restrict__ rw,
                              const float* __restrict__ rb, ushort_t* __restrict__ xcatT) {
  int t = threadIdx.x;
  int c = ((blockIdx.x & 1) << 8) + t;
  int b = blockIdx.x >> 1;
  float m = mu[b * CH + c], rs = rstd[b * CH + c];
  float w = rw[c], bb = rb[c];
  auto xn = [&](int j) {
    j = j < 0 ? 0 : (j > SEQL - 1 ? SEQL - 1 : j);
    return (x[(size_t)(b * SEQL + j) * CH + c] - m) * rs * w + bb;
  };
  float run = 0.f;
  for (int j = -12; j <= 12; ++j) run += xn(j);
  size_t ob = (size_t)(b * CH + c) * 1024;
  for (int l = 0; l < SEQL; ++l) {
    float mean = run * (1.f / 25.f);
    float v = xn(l);
    xcatT[ob + l]        = f2bf(v - mean);   // seasonal
    xcatT[ob + SEQL + l] = f2bf(mean);       // trend
    run += xn(l + 13) - xn(l - 12);
  }
}

// ---------------------------------------------------------------------------
// xout (B,C,E) fp32 -> bf16 (same layout) + bf16 transpose (B,E,C)
// ---------------------------------------------------------------------------
__global__ void trans_cvt_kernel(const float* __restrict__ in,
                                 ushort_t* __restrict__ o1, ushort_t* __restrict__ o2) {
  int id = blockIdx.x * 256 + threadIdx.x;  // B*512*512
  int e = id & 511;
  int c = (id >> 9) & 511;
  int b = id >> 18;
  float v = in[id];
  o1[id] = f2bf(v);
  o2[((size_t)(b * 512 + e) << 9) + c] = f2bf(v);
}

// ---------------------------------------------------------------------------
// Depthwise causal conv (k=4) + SiLU.  xz bf16 (B,L,2*DIN); xi = cols [0,DIN)
// ---------------------------------------------------------------------------
__global__ void conv_silu_kernel(const ushort_t* __restrict__ xz, const float* __restrict__ cw,
                                 const float* __restrict__ cb, ushort_t* __restrict__ xc) {
  int id = blockIdx.x * 256 + threadIdx.x;  // B*SEQL*DIN
  int d = id & (DIN - 1);
  int l = (id >> 10) & (SEQL - 1);
  int b = id >> 19;
  float acc = cb[d];
#pragma unroll
  for (int k = 0; k < 4; ++k) {
    int j = l + k - 3;
    if (j >= 0) acc += cw[d * 4 + k] * bf2f(xz[(size_t)(b * SEQL + j) * (2 * DIN) + d]);
  }
  xc[id] = f2bf(silu_f(acc));
}

// ---------------------------------------------------------------------------
// Selective scan. One workgroup per batch, 512 threads; each thread owns
// channels d and d+512 (16 states each, in registers).  Fuses  y+=D*u  and
// y *= silu(z); writes bf16 result in-place over xc (becomes out_proj input).
// ---------------------------------------------------------------------------
__global__ void __launch_bounds__(512) scan_kernel(
    const float* __restrict__ delta, ushort_t* __restrict__ xc,
    const ushort_t* __restrict__ xz, const float* __restrict__ dbl,
    const float* __restrict__ A_log, const float* __restrict__ Dp) {
  int b = blockIdx.x;
  int t = threadIdx.x;
  __shared__ float sm[32];                 // sm[0..15]=B_l, sm[16..31]=C_l
  float h0[NST], h1[NST], A0[NST], A1[NST];
  int d0 = t, d1 = t + 512;
#pragma unroll
  for (int s = 0; s < NST; ++s) {
    A0[s] = -__expf(A_log[d0 * NST + s]);
    A1[s] = -__expf(A_log[d1 * NST + s]);
    h0[s] = 0.f; h1[s] = 0.f;
  }
  float Dv0 = Dp[d0], Dv1 = Dp[d1];
  for (int l = 0; l < SEQL; ++l) {
    if (t < 32) sm[t] = dbl[(size_t)(b * SEQL + l) * 64 + 32 + t];
    __syncthreads();
    size_t rbase = (size_t)(b * SEQL + l) * DIN;
    size_t zbase = (size_t)(b * SEQL + l) * (2 * DIN) + DIN;
    {
      float dl = delta[rbase + d0];
      float u  = bf2f(xc[rbase + d0]);
      float du = dl * u, y = 0.f;
#pragma unroll
      for (int s = 0; s < NST; ++s) {
        float dA = __expf(dl * A0[s]);
        h0[s] = dA * h0[s] + du * sm[s];
        y += h0[s] * sm[16 + s];
      }
      float z = bf2f(xz[zbase + d0]);
      xc[rbase + d0] = f2bf((y + Dv0 * u) * silu_f(z));
    }
    {
      float dl = delta[rbase + d1];
      float u  = bf2f(xc[rbase + d1]);
      float du = dl * u, y = 0.f;
#pragma unroll
      for (int s = 0; s < NST; ++s) {
        float dA = __expf(dl * A1[s]);
        h1[s] = dA * h1[s] + du * sm[s];
        y += h1[s] * sm[16 + s];
      }
      float z = bf2f(xz[zbase + d1]);
      xc[rbase + d1] = f2bf((y + Dv1 * u) * silu_f(z));
    }
    __syncthreads();
  }
}

// ---------------------------------------------------------------------------
// xall = concat([x2^T, x1, x1+x2^T, xout], dim=-1) -> bf16 (B,512,2048)
// x2t is m2's output in (B,E,C); read transposed.
// ---------------------------------------------------------------------------
__global__ void concat_kernel(const float* __restrict__ x2t, const float* __restrict__ x1,
                              const float* __restrict__ x0, ushort_t* __restrict__ xall) {
  long long id = (long long)blockIdx.x * 256 + threadIdx.x;  // B*512*2048
  int col = (int)(id & 2047);
  int c   = (int)((id >> 11) & 511);
  int b   = (int)(id >> 20);
  int seg = col >> 9;
  int e   = col & 511;
  size_t ice = ((size_t)(b * 512 + c) << 9) + e;   // (b,c,e)
  size_t iec = ((size_t)(b * 512 + e) << 9) + c;   // (b,e,c)
  float v;
  if (seg == 0)      v = x2t[iec];
  else if (seg == 1) v = x1[ice];
  else if (seg == 2) v = x1[ice] + x2t[iec];
  else               v = x0[ice];
  xall[id] = f2bf(v);
}

// ---------------------------------------------------------------------------
// Final: transpose (B,C,96)->(B,96,C) + inverse RevIN
// ---------------------------------------------------------------------------
__global__ void final_kernel(const float* __restrict__ l3, const float* __restrict__ rw,
                             const float* __restrict__ rb, const float* __restrict__ mu,
                             const float* __restrict__ rstd, float* __restrict__ out) {
  int id = blockIdx.x * 256 + threadIdx.x;  // B*PRED*CH
  int c = id & 511;
  int p = (id >> 9) % PRED;
  int b = id / (PRED * CH);
  float o    = l3[((size_t)(b * 512 + c)) * PRED + p];
  float stdv = 1.f / rstd[b * CH + c];
  out[id] = (o - rb[c]) / (rw[c] + 1e-10f) * stdv + mu[b * CH + c];
}

// ---------------------------------------------------------------------------
// Host orchestration
// ---------------------------------------------------------------------------
extern "C" void kernel_launch(void* const* d_in, const int* in_sizes, int n_in,
                              void* d_out, int out_size, void* d_ws, size_t ws_size,
                              hipStream_t stream) {
  (void)in_sizes; (void)n_in; (void)out_size; (void)ws_size;
  const float* x     = (const float*)d_in[0];
  const float* revw  = (const float*)d_in[1];
  const float* revb  = (const float*)d_in[2];
  const float* lin2w = (const float*)d_in[3];
  const float* lin2b = (const float*)d_in[4];
  const float* lin3w = (const float*)d_in[5];
  const float* lin3b = (const float*)d_in[6];
  const float* m_inw[2]   = {(const float*)d_in[7],  (const float*)d_in[16]};
  const float* m_convw[2] = {(const float*)d_in[8],  (const float*)d_in[17]};
  const float* m_convb[2] = {(const float*)d_in[9],  (const float*)d_in[18]};
  const float* m_xw[2]    = {(const float*)d_in[10], (const float*)d_in[19]};
  const float* m_dtw[2]   = {(const float*)d_in[11], (const float*)d_in[20]};
  const float* m_dtb[2]   = {(const float*)d_in[12], (const float*)d_in[21]};
  const float* m_alog[2]  = {(const float*)d_in[13], (const float*)d_in[22]};
  const float* m_D[2]     = {(const float*)d_in[14], (const float*)d_in[23]};
  const float* m_outw[2]  = {(const float*)d_in[15], (const float*)d_in[24]};

  char* ws = (char*)d_ws;
  size_t off = 0;
  auto alloc = [&](size_t bytes) -> void* {
    void* p = ws + off;
    off += (bytes + 255) & ~(size_t)255;
    return p;
  };

  float*    mu_b    = (float*)   alloc((size_t)BATCH * CH * 4);
  float*    rstd_b  = (float*)   alloc((size_t)BATCH * CH * 4);
  ushort_t* xcatT   = (ushort_t*)alloc((size_t)BATCH * CH * 1024 * 2);
  ushort_t* wl2     = (ushort_t*)alloc((size_t)NEMBD * 1024 * 2);
  ushort_t* wl3     = (ushort_t*)alloc((size_t)PRED * 2048 * 2);
  ushort_t *winw[2], *wxw[2], *wdtw[2], *woutw[2];
  for (int i = 0; i < 2; ++i) {
    winw[i]  = (ushort_t*)alloc((size_t)2 * DIN * NEMBD * 2);
    wxw[i]   = (ushort_t*)alloc((size_t)64 * DIN * 2);
    wdtw[i]  = (ushort_t*)alloc((size_t)DIN * DTR * 2);
    woutw[i] = (ushort_t*)alloc((size_t)NEMBD * DIN * 2);
  }
  float*    xout     = (float*)   alloc((size_t)BATCH * CH * NEMBD * 4);
  ushort_t* xout_bf  = (ushort_t*)alloc((size_t)BATCH * CH * NEMBD * 2);
  ushort_t* xoutT_bf = (ushort_t*)alloc((size_t)BATCH * CH * NEMBD * 2);
  float*    x1       = (float*)   alloc((size_t)BATCH * CH * NEMBD * 4);
  float*    m2o      = (float*)   alloc((size_t)BATCH * CH * NEMBD * 4);
  ushort_t* xall     = (ushort_t*)alloc((size_t)BATCH * CH * 2048 * 2);
  float*    l3out    = (float*)   alloc((size_t)BATCH * CH * PRED * 4);
  // mamba scratch (reused by m1 then m2)
  ushort_t* xz    = (ushort_t*)alloc((size_t)BATCH * SEQL * 2 * DIN * 2);
  ushort_t* xcb   = (ushort_t*)alloc((size_t)BATCH * SEQL * DIN * 2);   // conv out, later ymul
  float*    dblf  = (float*)   alloc((size_t)BATCH * SEQL * 64 * 4);
  ushort_t* dblb  = (ushort_t*)alloc((size_t)BATCH * SEQL * 64 * 2);
  float*    delta = (float*)   alloc((size_t)BATCH * SEQL * DIN * 4);

  auto gemm = [&](const ushort_t* Ap, long long sA, int lda, const ushort_t* Wp,
                  void* Op, long long sO, int ldo, int M, int Nn, int Kk,
                  const float* bias, int postop, int obf) {
    dim3 g((Nn + TN - 1) / TN, M / TM, BATCH);
    gemm_bf16_kernel<<<g, dim3(256), 0, stream>>>(Ap, sA, lda, Wp, Op, sO, ldo,
                                                  Nn, Kk, bias, postop, obf);
  };
  auto cvt = [&](const float* in, ushort_t* out, int n) {
    cvt_kernel<<<(n + 255) / 256, 256, 0, stream>>>(in, out, n);
  };

  // --- weight conversion to bf16 ---
  cvt(lin2w, wl2, NEMBD * 1024);
  cvt(lin3w, wl3, PRED * 2048);
  for (int i = 0; i < 2; ++i) {
    cvt(m_inw[i],  winw[i],  2 * DIN * NEMBD);
    cvt(m_xw[i],   wxw[i],   64 * DIN);
    cvt(m_dtw[i],  wdtw[i],  DIN * DTR);
    cvt(m_outw[i], woutw[i], NEMBD * DIN);
  }

  // --- RevIN + decomposition ---
  stats_kernel<<<BATCH * 2, 256, 0, stream>>>(x, mu_b, rstd_b);
  decomp_kernel<<<BATCH * 2, 256, 0, stream>>>(x, mu_b, rstd_b, revw, revb, xcatT);

  // --- lin2: xout[b] (C x E) = xcatT[b] (C x 1024) * lin2_w^T + lin2_b ---
  gemm(xcatT, (long long)CH * 1024, 1024, wl2, xout, (long long)CH * NEMBD, NEMBD,
       CH, NEMBD, 1024, lin2b, 0, 0);
  trans_cvt_kernel<<<(BATCH * CH * NEMBD) / 256, 256, 0, stream>>>(xout, xout_bf, xoutT_bf);

  // --- two mamba blocks (m1 on (B,C,E), m2 on (B,E,C)) ---
  auto mamba = [&](int i, const ushort_t* xin, float* outp) {
    gemm(xin, (long long)SEQL * NEMBD, NEMBD, winw[i], xz,
         (long long)SEQL * 2 * DIN, 2 * DIN, SEQL, 2 * DIN, NEMBD, nullptr, 0, 1);
    conv_silu_kernel<<<(BATCH * SEQL * DIN) / 256, 256, 0, stream>>>(
        xz, m_convw[i], m_convb[i], xcb);
    gemm(xcb, (long long)SEQL * DIN, DIN, wxw[i], dblf,
         (long long)SEQL * 64, 64, SEQL, 64, DIN, nullptr, 0, 0);
    cvt_kernel<<<(BATCH * SEQL * 64) / 256, 256, 0, stream>>>(dblf, dblb, BATCH * SEQL * 64);
    gemm(dblb, (long long)SEQL * 64, 64, wdtw[i], delta,
         (long long)SEQL * DIN, DIN, SEQL, DIN, DTR, m_dtb[i], 1, 0);  // softplus fused
    scan_kernel<<<BATCH, 512, 0, stream>>>(delta, xcb, xz, dblf, m_alog[i], m_D[i]);
    gemm(xcb, (long long)SEQL * DIN, DIN, woutw[i], outp,
         (long long)SEQL * NEMBD, NEMBD, SEQL, NEMBD, DIN, nullptr, 0, 0);
  };
  mamba(0, xout_bf,  x1);
  mamba(1, xoutT_bf, m2o);

  // --- concat + lin3 + inverse RevIN ---
  concat_kernel<<<(BATCH * CH * 2048) / 256, 256, 0, stream>>>(m2o, x1, xout, xall);
  gemm(xall, (long long)CH * 2048, 2048, wl3, l3out, (long long)CH * PRED, PRED,
       CH, PRED, 2048, lin3b, 0, 0);
  final_kernel<<<(BATCH * PRED * CH) / 256, 256, 0, stream>>>(
      l3out, revw, revb, mu_b, rstd_b, (float*)d_out);
}
